// GraphEnergyModel_2911987827060
// MI455X (gfx1250) — matrix operand order repaired and checked by reading, not compiled, not run on hardware
//
#include <hip/hip_runtime.h>
#include <hip/hip_bf16.h>
#include <math.h>

typedef __attribute__((ext_vector_type(16))) _Float16 v16h;
typedef __attribute__((ext_vector_type(8)))  _Float16 v8h;
typedef __attribute__((ext_vector_type(8)))  float    v8f;

#define NUM_GRAPHS 1024
#define DIMH 128
#define K1 160              // 129 padded to multiple of 32
#define N1 256
#define K2 256
#define N2 256
#define K3 256
#define N3 64
#define STRIDE 264          // LDS row stride in f16 (16B-aligned: 264*2=528=33*16)
#define ROWS 48             // nodes per block (3 waves * 16)
#define BLOCK_THREADS 96    // 3 waves of 32

// f16 elements in workspace for each pre-swizzled weight
#define W1F_ELEMS (K1 * N1)          // 40960
#define W2F_ELEMS (K2 * N2)          // 65536
#define W3F_ELEMS (K3 * N3)          // 16384
#define PREP_TOTAL (W1F_ELEMS + W2F_ELEMS + W3F_ELEMS)  // 122880

// Fragment-ready offset for element (k,n) of a KxN weight matrix.
// B fragment in this code's contiguous-k order (same as A): lane = (k%32 >= 16)*16 + (n%16), the 16
// f16 values a lane owns for one 32x16 tile are contiguous (index k%16).
__device__ __forceinline__ size_t frag_off(int k, int n, int K, int N) {
    int ktiles = K >> 5;
    int kb32 = k >> 5, kr = k & 31;
    int nt = n >> 4, cl = n & 15;
    int lane = ((kr >> 4) << 4) | cl;
    int f = kr & 15;
    return ((size_t)(nt * ktiles + kb32) * 32 + lane) * 16 + f;
}

// Branchless GELU (tanh form) on HW transcendentals:
//   y  = 0.79788456*x + 0.03567741*x^3
//   t  = exp2(2*log2(e)*y)           (v_exp_f32)
//   gelu = x*t/(t+1) computed as x - x*rcp(t+1)   (v_rcp_f32; safe at t=0,inf)
__device__ __forceinline__ float fast_gelu(float x) {
    float x2 = x * x;
    float y  = x * __builtin_fmaf(0.0356774081f, x2, 0.7978845608f);
    float t  = __builtin_amdgcn_exp2f(y * 2.885390082f);
    float r  = __builtin_amdgcn_rcpf(t + 1.0f);
    return __builtin_fmaf(-x, r, x);
}

__global__ void zero_out_kernel(float* out, int n) {
    int i = blockIdx.x * blockDim.x + threadIdx.x;
    if (i < n) out[i] = 0.0f;
}

// Convert W1/W2/W3 to f16, reorder W1 rows ([x, h] -> [h, x, pad]),
// and swizzle into WMMA-B fragment order.
__global__ void prep_weights_kernel(const float* __restrict__ W1,
                                    const float* __restrict__ W2,
                                    const float* __restrict__ W3,
                                    _Float16* __restrict__ wf) {
    int t = blockIdx.x * blockDim.x + threadIdx.x;
    if (t < W1F_ELEMS) {
        int k = t >> 8, n = t & 255;        // K1 x 256 (padded-logical)
        float v;
        if (k < 128)       v = W1[(size_t)(k + 1) * 256 + n]; // h columns
        else if (k == 128) v = W1[n];                         // x column
        else               v = 0.0f;                          // pad rows
        wf[frag_off(k, n, K1, N1)] = (_Float16)v;
    } else if (t < W1F_ELEMS + W2F_ELEMS) {
        int u = t - W1F_ELEMS;
        int k = u >> 8, n = u & 255;
        (wf + W1F_ELEMS)[frag_off(k, n, K2, N2)] = (_Float16)W2[u];
    } else if (t < PREP_TOTAL) {
        int u = t - (W1F_ELEMS + W2F_ELEMS);
        int k = u >> 6, n = u & 63;
        (wf + W1F_ELEMS + W2F_ELEMS)[frag_off(k, n, K3, N3)] = (_Float16)W3[u];
    }
}

// One fused MLP layer for this wave's 16 rows: out = gelu(in @ W + b).
__device__ __forceinline__ void layer_wmma(const _Float16* __restrict__ inBuf,
                                           _Float16* __restrict__ outBuf,
                                           const _Float16* __restrict__ wfrag,
                                           const float* __restrict__ bias,
                                           int K, int Nout,
                                           int wave, int lane, bool applyGelu) {
    const int ktiles = K >> 5;
    const int m0    = wave << 4;
    const int row   = m0 + (lane & 15);
    const int khalf = (lane >> 4) << 4;   // 0 or 16: A-frag K sub-block
    const int cl    = lane & 15;          // output column within tile
    const int mhi   = (lane >> 4) << 3;   // +8 for lanes 16..31 on C rows
    const _Float16* __restrict__ rowPtr = inBuf + (size_t)row * STRIDE + khalf;

    for (int nt = 0; nt < (Nout >> 4); ++nt) {
        float bval = bias[(nt << 4) + cl];
        v8f acc;
#pragma unroll
        for (int r = 0; r < 8; ++r) acc[r] = bval;

        for (int kb32 = 0; kb32 < ktiles; ++kb32) {
            // A fragment: two contiguous 16B runs from LDS
            const _Float16* ap = rowPtr + (kb32 << 5);
            v8h lo = *(const v8h*)(ap);
            v8h hi = *(const v8h*)(ap + 8);
            v16h a;
#pragma unroll
            for (int i = 0; i < 8; ++i) { a[i] = lo[i]; a[8 + i] = hi[i]; }
            // B fragment: one contiguous 32B run from pre-swizzled global
            v16h b = *(const v16h*)(wfrag +
                        (((size_t)(nt * ktiles + kb32) << 5) + lane) * 16);
            acc = __builtin_amdgcn_wmma_f32_16x16x32_f16(
                      false, a, false, b, (short)0, acc, false, false);
        }
#pragma unroll
        for (int r = 0; r < 8; ++r) {
            float v = acc[r];
            float y = applyGelu ? fast_gelu(v) : v;
            outBuf[(size_t)(m0 + mhi + r) * STRIDE + (nt << 4) + cl] = (_Float16)y;
        }
    }
}

__global__ __launch_bounds__(BLOCK_THREADS)
void fused_mlp_kernel(const float* __restrict__ x,
                      const float* __restrict__ h,
                      const int*   __restrict__ bidx,
                      const float* __restrict__ b1,
                      const float* __restrict__ b2,
                      const float* __restrict__ b3,
                      const float* __restrict__ W4,
                      const float* __restrict__ b4,
                      const _Float16* __restrict__ wf,
                      float* __restrict__ out,
                      int N) {
    __shared__ _Float16 bufA[ROWS * STRIDE];
    __shared__ _Float16 bufB[ROWS * STRIDE];
    __shared__ int sidx[ROWS];

    const int tid  = threadIdx.x;
    const int wave = tid >> 5;
    const int lane = tid & 31;
    const long node0 = (long)blockIdx.x * ROWS;

    // Stage batch indices
    for (int i = tid; i < ROWS; i += BLOCK_THREADS) {
        long node = node0 + i;
        sidx[i] = (node < N) ? bidx[node] : 0;
    }
    __syncthreads();

    // Stage combined = [h[g], x, pad] as f16 into bufA (cols 0..159)
    for (int t = tid; t < ROWS * DIMH; t += BLOCK_THREADS) {
        int r = t >> 7, c = t & (DIMH - 1);
        long node = node0 + r;
        float v = (node < N) ? h[(size_t)sidx[r] * DIMH + c] : 0.0f;
        bufA[(size_t)r * STRIDE + c] = (_Float16)v;
    }
    for (int t = tid; t < ROWS * 32; t += BLOCK_THREADS) {
        int r = t >> 5, c = DIMH + (t & 31);   // cols 128..159
        long node = node0 + r;
        float v = (c == DIMH && node < N) ? x[node] : 0.0f;
        bufA[(size_t)r * STRIDE + c] = (_Float16)v;
    }
    __syncthreads();

    const _Float16* wf1 = wf;
    const _Float16* wf2 = wf + W1F_ELEMS;
    const _Float16* wf3 = wf + W1F_ELEMS + W2F_ELEMS;

    layer_wmma(bufA, bufB, wf1, b1, K1, N1, wave, lane, true);   // 129->256
    __syncthreads();
    layer_wmma(bufB, bufA, wf2, b2, K2, N2, wave, lane, true);   // 256->256
    __syncthreads();
    layer_wmma(bufA, bufB, wf3, b3, K3, N3, wave, lane, true);   // 256->64
    __syncthreads();

    // Layer 4: 64->1 scalar dot + segment-sum via f32 global atomics
    if (tid < ROWS) {
        long node = node0 + tid;
        if (node < N) {
            float e = b4[0];
#pragma unroll
            for (int k = 0; k < N3; ++k)
                e += (float)bufB[(size_t)tid * STRIDE + k] * W4[k];
            __hip_atomic_fetch_add(&out[sidx[tid]], e,
                                   __ATOMIC_RELAXED, __HIP_MEMORY_SCOPE_AGENT);
        }
    }
}

extern "C" void kernel_launch(void* const* d_in, const int* in_sizes, int n_in,
                              void* d_out, int out_size, void* d_ws, size_t ws_size,
                              hipStream_t stream) {
    const float* x    = (const float*)d_in[0];
    const float* h    = (const float*)d_in[1];
    const int*   bidx = (const int*)  d_in[2];
    const float* W1   = (const float*)d_in[3];
    const float* b1   = (const float*)d_in[4];
    const float* W2   = (const float*)d_in[5];
    const float* b2   = (const float*)d_in[6];
    const float* W3   = (const float*)d_in[7];
    const float* b3   = (const float*)d_in[8];
    const float* W4   = (const float*)d_in[9];
    const float* b4   = (const float*)d_in[10];
    float* out = (float*)d_out;
    _Float16* wf = (_Float16*)d_ws;
    const int N = in_sizes[0];

    zero_out_kernel<<<(out_size + 255) / 256, 256, 0, stream>>>(out, out_size);
    prep_weights_kernel<<<(PREP_TOTAL + 255) / 256, 256, 0, stream>>>(W1, W2, W3, wf);
    const int blocks = (N + ROWS - 1) / ROWS;
    fused_mlp_kernel<<<blocks, BLOCK_THREADS, 0, stream>>>(
        x, h, bidx, b1, b2, b3, W4, b4, wf, out, N);
}
